// Model_62045097558368
// MI455X (gfx1250) — matrix operand order repaired and checked by reading, not compiled, not run on hardware
//
#include <hip/hip_runtime.h>

// ---------------------------------------------------------------------------
// Model: embed -> conv1d(win=3 over time, full E) -> biGRU(H=128) ->
//        masked mean over time -> linear -> sigmoid
// Dims: V=50000 E=64 B=64 T=1024 F=128 H=128 WIN=3 PAD=1
//   K1: im2col GEMM (M=B*T=65536, N=F=128, K=192) via wmma bf16 -> feats(bf16)
//   K2: 8 persistent WGs (2 dirs x 4 batch-chunks of 16). bf16 weights resident
//       in LDS (2 x 96KB). Per step: async-prefetch next feature tile to LDS
//       (global_load_async_to_lds_b128, ASYNCcnt), dual GEMM with
//       v_wmma_f32_16x16x32_bf16, GRU gates fused IN REGISTERS (C-tile columns
//       {h,128+h,256+h} land on the same lane), h-state fp32 in registers,
//       only bf16 h copy goes through LDS for the next A operand.
//   K3: tiny head.
// ---------------------------------------------------------------------------

typedef __attribute__((ext_vector_type(16))) __bf16 v16bf;
typedef __attribute__((ext_vector_type(8)))  float  v8f;

union ABFrag { uint4 u[2]; v16bf v; };

// A fragment: 16x32 bf16 tile, row-major in LDS, 'stride' elems/row.
// lanes 0-15 rows M hold K={0..7,16..23}; lanes 16-31 hold K={8..15,24..31}.
__device__ inline v16bf load_a16x32(const __bf16* tile, int stride, int kbase) {
  int lane = threadIdx.x & 31;
  int m = lane & 15, half = lane >> 4;
  const __bf16* p = tile + m * stride + kbase + 8 * half;
  ABFrag f;
  f.u[0] = *(const uint4*)(p);
  f.u[1] = *(const uint4*)(p + 16);
  return f.v;
}

// B fragment: 32x16 bf16 (KxN), stored N-major: bt[n][k], 'stride' = K dim.
// lanes 0-15 col N hold K={0..7,16..23}; lanes 16-31 hold K={8..15,24..31}.
__device__ inline v16bf load_b32x16(const __bf16* bt, int stride, int nbase, int kbase) {
  int lane = threadIdx.x & 31;
  int n = lane & 15, half = lane >> 4;
  const __bf16* p = bt + (nbase + n) * stride + kbase + 8 * half;
  ABFrag f;
  f.u[0] = *(const uint4*)(p);
  f.u[1] = *(const uint4*)(p + 16);
  return f.v;
}

__device__ inline v8f wmma_bf16(v16bf a, v16bf b, v8f c) {
  return __builtin_amdgcn_wmma_f32_16x16x32_bf16(false, a, false, b, (short)0, c,
                                                 false, false);
}

__device__ inline float sigmoidf_(float x) { return 1.f / (1.f + __expf(-x)); }

// Async copy 16B global -> LDS, tracked by ASYNCcnt (CDNA5).
__device__ inline void async_g2l_b128(void* lds_dst, const void* gsrc) {
  unsigned int  la = (unsigned int)(unsigned long long)lds_dst;   // addr[31:0] = LDS addr
  unsigned long long ga = (unsigned long long)gsrc;
  asm volatile("global_load_async_to_lds_b128 %0, %1, off"
               :: "v"(la), "v"(ga) : "memory");
}
__device__ inline void wait_asynccnt0() {
  asm volatile("s_wait_asynccnt 0x0" ::: "memory");
}

// ---------------------------------------------------------------------------
// K1: embedding gather + conv-as-GEMM.  feats[m=b*T+t][f] bf16.
// Grid: 512 WGs x 256 thr. Each WG: 128 rows x 128 cols, K=192.
// ---------------------------------------------------------------------------
__global__ void embed_conv_kernel(const int* __restrict__ ipts,
                                  const float* __restrict__ emb,
                                  const float* __restrict__ conv_w,
                                  const float* __restrict__ conv_b,
                                  __bf16* __restrict__ feats) {
  extern __shared__ char smem[];
  __bf16* Atile = (__bf16*)smem;                       // 128 x 192 bf16
  __bf16* Bt    = (__bf16*)(smem + 128 * 192 * 2);     // 128 x 192 bf16 n-major
  float*  bias  = (float*)(smem + 2 * 128 * 192 * 2);  // 128 f32

  const int tid = threadIdx.x;
  const int rowbase = blockIdx.x * 128;

  for (int i = tid; i < 128 * 192; i += 256) Bt[i] = (__bf16)conv_w[i];
  if (tid < 128) bias[tid] = conv_b[tid];

  for (int i = tid; i < 128 * 192; i += 256) {
    int r = i / 192, c = i - r * 192;
    int m = rowbase + r;
    int b = m >> 10, t = m & 1023;
    int w = c >> 6, e = c & 63;
    int ts = t + w - 1;
    float v = 0.f;
    if (ts >= 0 && ts < 1024) {
      int tok = ipts[(b << 10) + ts];
      v = emb[(size_t)tok * 64 + e];
    }
    Atile[i] = (__bf16)v;
  }
  __syncthreads();

  const int wave = tid >> 5;
  const __bf16* Arow = Atile + wave * 16 * 192;
  v8f acc[8];
  for (int nt = 0; nt < 8; ++nt) acc[nt] = (v8f){0, 0, 0, 0, 0, 0, 0, 0};

  for (int kk = 0; kk < 6; ++kk) {
    v16bf a = load_a16x32(Arow, 192, kk * 32);
    for (int nt = 0; nt < 8; ++nt) {
      v16bf b = load_b32x16(Bt, 192, nt * 16, kk * 32);
      acc[nt] = wmma_bf16(a, b, acc[nt]);
    }
  }

  int lane = tid & 31, n = lane & 15, half = lane >> 4;
  for (int nt = 0; nt < 8; ++nt) {
    int col = nt * 16 + n;
    float bv = bias[col];
    for (int r = 0; r < 8; ++r) {
      int m = rowbase + wave * 16 + half * 8 + r;
      feats[(size_t)m * 128 + col] = (__bf16)(acc[nt][r] + bv);
    }
  }
}

// ---------------------------------------------------------------------------
// K2: GRU scan. Grid: 8 WGs x 256 thr. WG = (dir, 16-row batch chunk).
// ---------------------------------------------------------------------------
#define SM_WIH   0                    // 384x128 bf16 = 98304
#define SM_WHH   98304                // 384x128 bf16 = 98304
#define SM_HBF   196608               // 16x128 bf16  = 4096
#define SM_FB0   200704               // 2 x 16x128 bf16 = 2 x 4096 (ping/pong)
#define SM_LEN   208896               // 16 int
#define SM_TOTAL 208960

__global__ void gru_scan_kernel(const __bf16* __restrict__ feats,
                                const int* __restrict__ seq_lengths,
                                const float* __restrict__ hidden,
                                const float* __restrict__ w_ih_f,
                                const float* __restrict__ w_hh_f,
                                const float* __restrict__ b_ih_f,
                                const float* __restrict__ b_hh_f,
                                const float* __restrict__ w_ih_b,
                                const float* __restrict__ w_hh_b,
                                const float* __restrict__ b_ih_b,
                                const float* __restrict__ b_hh_b,
                                float* __restrict__ mot) {
  extern __shared__ char smem[];
  __bf16* Wih = (__bf16*)(smem + SM_WIH);
  __bf16* Whh = (__bf16*)(smem + SM_WHH);
  __bf16* hbf = (__bf16*)(smem + SM_HBF);
  int*    len = (int*)(smem + SM_LEN);

  const int tid  = threadIdx.x;
  const int dir  = blockIdx.x >> 2;
  const int b0   = (blockIdx.x & 3) * 16;

  const float* wih = dir ? w_ih_b : w_ih_f;
  const float* whh = dir ? w_hh_b : w_hh_f;
  const float* vih = dir ? b_ih_b : b_ih_f;
  const float* vhh = dir ? b_hh_b : b_hh_f;

  const int wave = tid >> 5;
  const int lane = tid & 31;
  const int cn = lane & 15, chalf = lane >> 4;
  const int hcol = wave * 16 + cn;            // this lane's hidden column
  const int rrow = chalf * 8;                 // row base (rows rrow..rrow+7)

  // Stage weights (PyTorch (3H,in) layout is already N-major = Bt layout).
  for (int i = tid; i < 384 * 128; i += 256) {
    Wih[i] = (__bf16)wih[i];
    Whh[i] = (__bf16)whh[i];
  }
  if (tid < 16) len[tid] = seq_lengths[b0 + tid];

  // Per-lane register state: biases, h (fp32), masked sums, lengths.
  float bx[3], bh[3];
  for (int q = 0; q < 3; ++q) {
    int col = (wave + q * 8) * 16 + cn;
    bx[q] = vih[col];
    bh[q] = vhh[col];
  }
  float hreg[8], s[8];
  int lenr[8];
  for (int r = 0; r < 8; ++r) {
    hreg[r] = hidden[(size_t)dir * 64 * 128 + (size_t)(b0 + rrow + r) * 128 + hcol];
    s[r] = 0.f;
    hbf[(rrow + r) * 128 + hcol] = (__bf16)hreg[r];
  }
  __syncthreads();
  for (int r = 0; r < 8; ++r) lenr[r] = len[rrow + r];

  // Async staging assignment: thread -> (row, 16B chunk).
  const int srow = tid >> 4;
  const int schk = (tid & 15) * 8;            // bf16 offset within row
  const int slen = len[srow];
  const size_t srcrow = ((size_t)(b0 + srow)) << 10;  // feats row base (in T units)
  const int sdst = srow * 128 + schk;         // bf16 offset within a buffer

  // Prologue: async-stage step 0 into buffer 0.
  {
    int ts = dir ? (slen - 1 < 0 ? 0 : slen - 1) : 0;
    async_g2l_b128((__bf16*)(smem + SM_FB0) + sdst,
                   feats + (srcrow + (unsigned)ts) * 128 + schk);
  }

  for (int t = 0; t < 1024; ++t) {
    wait_asynccnt0();      // this wave's prefetch for buffer (t&1) has landed
    __syncthreads();       // everyone's prefetch landed; hbf(t) visible

    // Issue async prefetch of step t+1 into the other buffer.
    if (t < 1023) {
      int tn = t + 1;
      int ts = dir ? (slen - 1 - tn < 0 ? 0 : slen - 1 - tn) : tn;
      __bf16* nbuf = (__bf16*)(smem + SM_FB0 + ((tn & 1) * 4096));
      async_g2l_b128(nbuf + sdst,
                     feats + (srcrow + (unsigned)ts) * 128 + schk);
    }

    // Dual GEMM, A-fragments shared across this wave's 3 N-tiles.
    const __bf16* fcur = (const __bf16*)(smem + SM_FB0 + ((t & 1) * 4096));
    v8f ax[3], ah[3];
    for (int q = 0; q < 3; ++q) {
      ax[q] = (v8f){0, 0, 0, 0, 0, 0, 0, 0};
      ah[q] = (v8f){0, 0, 0, 0, 0, 0, 0, 0};
    }
#pragma unroll
    for (int kk = 0; kk < 4; ++kk) {
      v16bf aF = load_a16x32(fcur, 128, kk * 32);
      v16bf aH = load_a16x32(hbf,  128, kk * 32);
#pragma unroll
      for (int q = 0; q < 3; ++q) {
        int nb = (wave + q * 8) * 16;
        v16bf bX = load_b32x16(Wih, 128, nb, kk * 32);
        ax[q] = wmma_bf16(aF, bX, ax[q]);
        v16bf bH = load_b32x16(Whh, 128, nb, kk * 32);
        ah[q] = wmma_bf16(aH, bH, ah[q]);
      }
    }

    // GRU gates fused in registers: q=0 -> r-gate, q=1 -> z-gate, q=2 -> n-gate,
    // all for column hcol, rows rrow..rrow+7 (C-tile layout).
    float hnew[8];
#pragma unroll
    for (int r = 0; r < 8; ++r) {
      float rg = sigmoidf_((ax[0][r] + bx[0]) + (ah[0][r] + bh[0]));
      float zg = sigmoidf_((ax[1][r] + bx[1]) + (ah[1][r] + bh[1]));
      float nn = tanhf((ax[2][r] + bx[2]) + rg * (ah[2][r] + bh[2]));
      float hn = (1.f - zg) * nn + zg * hreg[r];
      hreg[r] = hn;
      hnew[r] = hn;
      if (t < lenr[r]) s[r] += hn;
    }

    __syncthreads();       // all waves done reading hbf(t)
#pragma unroll
    for (int r = 0; r < 8; ++r)
      hbf[(rrow + r) * 128 + hcol] = (__bf16)hnew[r];
    // next iteration's top barrier publishes hbf(t+1)
  }

  // Masked mean -> mot[dir*64 + b][h]
  for (int r = 0; r < 8; ++r)
    mot[(size_t)(dir * 64 + b0 + rrow + r) * 128 + hcol] = s[r] / (float)lenr[r];
}

// ---------------------------------------------------------------------------
// K3: head. out[b] = sigmoid(mot_f . w[:128] + mot_b . w[128:] + b)
// ---------------------------------------------------------------------------
__global__ void head_kernel(const float* __restrict__ mot,
                            const float* __restrict__ lin_w,
                            const float* __restrict__ lin_b,
                            float* __restrict__ out) {
  int b = threadIdx.x;
  float acc = lin_b[0];
  for (int h = 0; h < 128; ++h) acc += mot[(size_t)b * 128 + h] * lin_w[h];
  for (int h = 0; h < 128; ++h) acc += mot[(size_t)(64 + b) * 128 + h] * lin_w[128 + h];
  out[b] = sigmoidf_(acc);
}

// ---------------------------------------------------------------------------
extern "C" void kernel_launch(void* const* d_in, const int* in_sizes, int n_in,
                              void* d_out, int out_size, void* d_ws, size_t ws_size,
                              hipStream_t stream) {
  const int*   ipts    = (const int*)d_in[0];
  const int*   lens    = (const int*)d_in[1];
  const float* hidden  = (const float*)d_in[2];
  const float* emb     = (const float*)d_in[3];
  const float* conv_w  = (const float*)d_in[4];
  const float* conv_b  = (const float*)d_in[5];
  const float* w_ih_f  = (const float*)d_in[6];
  const float* w_hh_f  = (const float*)d_in[7];
  const float* b_ih_f  = (const float*)d_in[8];
  const float* b_hh_f  = (const float*)d_in[9];
  const float* w_ih_b  = (const float*)d_in[10];
  const float* w_hh_b  = (const float*)d_in[11];
  const float* b_ih_b  = (const float*)d_in[12];
  const float* b_hh_b  = (const float*)d_in[13];
  const float* lin_w   = (const float*)d_in[14];
  const float* lin_b   = (const float*)d_in[15];
  float* out = (float*)d_out;

  __bf16* feats = (__bf16*)d_ws;                                   // 16MB
  float*  mot   = (float*)((char*)d_ws + (size_t)64 * 1024 * 128 * 2);

  size_t smem1 = 2 * 128 * 192 * sizeof(__bf16) + 128 * sizeof(float);
  embed_conv_kernel<<<512, 256, smem1, stream>>>(ipts, emb, conv_w, conv_b, feats);

  gru_scan_kernel<<<8, 256, SM_TOTAL, stream>>>(feats, lens, hidden,
                                                w_ih_f, w_hh_f, b_ih_f, b_hh_f,
                                                w_ih_b, w_hh_b, b_ih_b, b_hh_b,
                                                mot);

  head_kernel<<<1, 64, 0, stream>>>(mot, lin_w, lin_b, out);
}
